// Qwen3TTSTokenizerSingleCodebookWhisperEncoderVQ_12524124636039
// MI455X (gfx1250) — matrix operand order repaired and checked by reading, not compiled, not run on hardware
//
#include <hip/hip_runtime.h>
#include <hip/hip_bf16.h>
#include <stdint.h>

typedef __attribute__((ext_vector_type(16))) __bf16 v16bf;
typedef __attribute__((ext_vector_type(8)))  float  v8f;

namespace {
constexpr int T_   = 65536;
constexpr int G_   = 2;
constexpr int NQ_  = 4;
constexpr int K_   = 2048;
constexpr int D_   = 256;
constexpr int ROWS = 128;            // t-rows per workgroup (8 waves x 16)
constexpr int RSTRIDE = 260;         // padded f32 residual row stride (LDS)
constexpr int ESTRIDE = 264;         // padded bf16 codebook row stride (528B, 16B aligned)
constexpr int KT   = K_ / 16;        // 128 code tiles of 16
constexpr size_t R_BYTES = (size_t)ROWS * RSTRIDE * 4;   // 133120
constexpr size_t E_BYTES = (size_t)2 * 16 * ESTRIDE * 2; // 16896 (double buffer)
constexpr size_t I_BYTES = 8 * 16 * 4;                   // per-wave argmin scratch
constexpr size_t LDS_BYTES = R_BYTES + E_BYTES + I_BYTES;
}

// ---------------- pre-pass: f32 codebook -> bf16 + row norms ----------------
__global__ __launch_bounds__(256) void vq_prep(const float* __restrict__ embed,
                                               __bf16* __restrict__ ebf,
                                               float* __restrict__ norms) {
  const int row  = blockIdx.x * 8 + (threadIdx.x >> 5);   // G*NQ*K rows
  const int lane = threadIdx.x & 31;
  const float* src = embed + (size_t)row * D_;
  __bf16*      dst = ebf   + (size_t)row * D_;
  float ss = 0.f;
#pragma unroll
  for (int i = 0; i < 8; ++i) {
    float v = src[lane + i * 32];
    ss += v * v;
    dst[lane + i * 32] = (__bf16)v;
  }
#pragma unroll
  for (int m = 16; m >= 1; m >>= 1) ss += __shfl_xor(ss, m, 32);
  if (lane == 0) norms[row] = ss;
}

// ------------- async stage one 16x256 bf16 codebook tile into LDS ----------
__device__ __forceinline__ void issue_tile(const __bf16* __restrict__ src,
                                           uint32_t lds_base, int tid) {
  // 16 rows x 512B source -> LDS rows of 528B; 256 threads x 2 x 16B chunks
#pragma unroll
  for (int i = 0; i < 2; ++i) {
    int c = tid * 2 + i;               // 0..511
    int r = c >> 5;                    // row 0..15
    int s = c & 31;                    // 16B segment
    uint32_t voff = (uint32_t)(r * 512 + s * 16);
    uint32_t ldso = lds_base + (uint32_t)(r * (ESTRIDE * 2) + s * 16);
    asm volatile("global_load_async_to_lds_b128 %0, %1, %2"
                 :: "v"(ldso), "v"(voff), "s"(src) : "memory");
  }
}

// ------------------------------- main kernel -------------------------------
__global__ __launch_bounds__(256, 1) void vq_main(const float* __restrict__ x,
                                                  const float* __restrict__ embed,
                                                  const __bf16* __restrict__ ebf,
                                                  const float* __restrict__ norms,
                                                  int* __restrict__ outIdx,
                                                  float* __restrict__ outQ) {
  extern __shared__ char smem[];
  float*  rlds = (float*)smem;                        // [ROWS][RSTRIDE] f32 residual
  __bf16* ebuf = (__bf16*)(smem + R_BYTES);           // 2 x [16][ESTRIDE] bf16
  int*    idxs = (int*)(smem + R_BYTES + E_BYTES);    // [8 waves][16 rows]

  const int g    = blockIdx.y;
  const int t0   = blockIdx.x * ROWS;
  const int tid  = threadIdx.x;
  const int w    = tid >> 5;
  const int lane = tid & 31;
  const int m    = lane & 15;    // WMMA row (A) / column (B,C)
  const int half = lane >> 4;
  const int row  = w * 16 + m;   // t-row within block for compute phase

  const uint32_t ebuf_lds0 = (uint32_t)(uintptr_t)(ebuf);
  const uint32_t ebuf_lds1 = (uint32_t)(uintptr_t)(ebuf + 16 * ESTRIDE);

  // ---- load x tile into LDS residual (each thread: half a row, 128 f32) ----
  {
    const int lr = tid >> 1, hh = tid & 1;
    const float4* src = (const float4*)(x + (size_t)(t0 + lr) * (G_ * D_) + g * D_ + hh * 128);
    float4* dst = (float4*)(rlds + lr * RSTRIDE + hh * 128);
#pragma unroll
    for (int i = 0; i < 32; ++i) dst[i] = src[i];
  }
  __syncthreads();

  for (int q = 0; q < NQ_; ++q) {
    // ---- convert my residual rows to the bf16 A operand (kept in VGPRs) ----
    v16bf abf[8];
    {
      const float* rp = rlds + row * RSTRIDE;
#pragma unroll
      for (int c = 0; c < 8; ++c) {
#pragma unroll
        for (int j = 0; j < 8; ++j) {
          abf[c][j]     = (__bf16)rp[c * 32 + half * 8 + j];
          abf[c][j + 8] = (__bf16)rp[c * 32 + 16 + half * 8 + j];
        }
      }
    }

    const __bf16* eq = ebf   + (size_t)(g * NQ_ + q) * K_ * D_;
    const float*  nq = norms + (size_t)(g * NQ_ + q) * K_;

    issue_tile(eq, ebuf_lds0, tid);   // prefetch tile 0

    float bestD[8];
    int   bestI[8];
#pragma unroll
    for (int v = 0; v < 8; ++v) { bestD[v] = 3.4e38f; bestI[v] = 0; }

    for (int kt = 0; kt < KT; ++kt) {
      asm volatile("s_wait_asynccnt 0" ::: "memory");
      __syncthreads();
      if (kt + 1 < KT)
        issue_tile(eq + (size_t)(kt + 1) * 16 * D_,
                   ((kt + 1) & 1) ? ebuf_lds1 : ebuf_lds0, tid);

      const __bf16* eb = ebuf + ((kt & 1) ? 16 * ESTRIDE : 0);

      // Load ALL 8 B-operand chunks first (16 ds_load_b128, single wait),
      // then run the 8 WMMAs back-to-back on two accumulator chains.
      union BB { uint4 u[2]; v16bf v; } bb[8];
      {
        const uint4* bp = (const uint4*)(eb + m * ESTRIDE + half * 8);
#pragma unroll
        for (int c = 0; c < 8; ++c) {
          bb[c].u[0] = bp[c * 4];       // +c*64 bytes
          bb[c].u[1] = bp[c * 4 + 2];   // +c*64+32 bytes
        }
      }
      v8f acc0 = {0.f, 0.f, 0.f, 0.f, 0.f, 0.f, 0.f, 0.f};
      v8f acc1 = {0.f, 0.f, 0.f, 0.f, 0.f, 0.f, 0.f, 0.f};
#pragma unroll
      for (int c = 0; c < 8; c += 2) {
        acc0 = __builtin_amdgcn_wmma_f32_16x16x32_bf16(
            false, abf[c],     false, bb[c].v,     (short)0, acc0, false, false);
        acc1 = __builtin_amdgcn_wmma_f32_16x16x32_bf16(
            false, abf[c + 1], false, bb[c + 1].v, (short)0, acc1, false, false);
      }

      const float nrm = nq[kt * 16 + m];
      const int   ki  = kt * 16 + m;
#pragma unroll
      for (int v = 0; v < 8; ++v) {
        float dv = nrm - 2.0f * (acc0[v] + acc1[v]);
        if (dv < bestD[v] || (dv == bestD[v] && ki < bestI[v])) {
          bestD[v] = dv; bestI[v] = ki;
        }
      }
    }

    // ---- argmin across the 16 code-columns of each lane half ----
#pragma unroll
    for (int v = 0; v < 8; ++v) {
      float d0 = bestD[v]; int i0 = bestI[v];
#pragma unroll
      for (int sh = 1; sh < 16; sh <<= 1) {
        float d1 = __shfl_xor(d0, sh, 32);
        int   i1 = __shfl_xor(i0, sh, 32);
        if (d1 < d0 || (d1 == d0 && i1 < i0)) { d0 = d1; i0 = i1; }
      }
      bestI[v] = i0;   // uniform within each half; half h holds rows 8h+v
    }
    if (lane == 0) {
#pragma unroll
      for (int v = 0; v < 8; ++v) idxs[w * 16 + v] = bestI[v];
    }
    if (lane == 16) {
#pragma unroll
      for (int v = 0; v < 8; ++v) idxs[w * 16 + 8 + v] = bestI[v];
    }
    asm volatile("s_wait_dscnt 0" ::: "memory");

    const int myIdx = idxs[w * 16 + m];
    if (half == 0)
      outIdx[((size_t)q * G_ + g) * T_ + t0 + w * 16 + m] = myIdx;

    // ---- gather winning f32 code row (L2-resident) and update residual ----
    {
      const float* ev = embed + ((size_t)(g * NQ_ + q) * K_ + myIdx) * D_;
      float* rp = rlds + row * RSTRIDE;
#pragma unroll
      for (int c = 0; c < 8; ++c) {
#pragma unroll
        for (int j = 0; j < 8; ++j) {
          rp[c * 32 + half * 8 + j]      -= ev[c * 32 + half * 8 + j];
          rp[c * 32 + 16 + half * 8 + j] -= ev[c * 32 + 16 + half * 8 + j];
        }
      }
    }
  }

  __syncthreads();
  // ---- quant = x - r_final ----
  {
    const int lr = tid >> 1, hh = tid & 1;
    const float* xs = x    + (size_t)(t0 + lr) * (G_ * D_) + g * D_ + hh * 128;
    const float* rs = rlds + lr * RSTRIDE + hh * 128;
    float*       qd = outQ + (size_t)(t0 + lr) * (G_ * D_) + g * D_ + hh * 128;
#pragma unroll
    for (int i = 0; i < 128; i += 4) {
      float4 xv = *(const float4*)(xs + i);
      float4 rv = *(const float4*)(rs + i);
      float4 ov = {xv.x - rv.x, xv.y - rv.y, xv.z - rv.z, xv.w - rv.w};
      *(float4*)(qd + i) = ov;
    }
  }
}

extern "C" void kernel_launch(void* const* d_in, const int* in_sizes, int n_in,
                              void* d_out, int out_size, void* d_ws, size_t ws_size,
                              hipStream_t stream) {
  const float* x     = (const float*)d_in[0];   // [T, G*D]
  const float* embed = (const float*)d_in[1];   // [G, NQ, K, D]

  __bf16* ebf   = (__bf16*)d_ws;                                  // 8 MB
  float*  norms = (float*)((char*)d_ws + (size_t)G_ * NQ_ * K_ * D_ * 2);

  const int rows = G_ * NQ_ * K_;               // 16384 codebook rows
  vq_prep<<<rows / 8, 256, 0, stream>>>(embed, ebf, norms);

  int*   outIdx = (int*)d_out;                              // [NQ, G, T] int32
  float* outQ   = (float*)d_out + (size_t)NQ_ * G_ * T_;    // [T, G*D] f32

  dim3 grid(T_ / ROWS, G_);
  vq_main<<<grid, 256, (uint32_t)LDS_BYTES, stream>>>(x, embed, ebf, norms,
                                                      outIdx, outQ);
}
